// AttentionBlock_34531537060191
// MI455X (gfx1250) — matrix operand order repaired and checked by reading, not compiled, not run on hardware
//
#include <hip/hip_runtime.h>
#include <hip/hip_bf16.h>

// ---------------------------------------------------------------------------
// Transformer block (pre-LN attn + FFN) for MI455X (gfx1250), wave32 + WMMA.
// GEMMs: v_wmma_f32_16x16x32_bf16 (bf16 operands, fp32 accumulate).
// Flash-attn: block-cooperative K/V staging via async global->LDS copies
// (ASYNCcnt pipeline, double buffered), causal online softmax.
// ---------------------------------------------------------------------------

typedef __bf16 bf16;
typedef __attribute__((ext_vector_type(16))) __bf16 bf16x16;
typedef __attribute__((ext_vector_type(8)))  __bf16 bf16x8;
typedef __attribute__((ext_vector_type(4)))  __bf16 bf16x4;
typedef __attribute__((ext_vector_type(8)))  float  f32x8;
typedef __attribute__((ext_vector_type(4)))  int    int4v;
typedef __attribute__((address_space(1))) int4v* as1_int4p;  // global
typedef __attribute__((address_space(3))) int4v* as3_int4p;  // LDS

constexpr int Bz = 2, Tz = 2048, Ez = 1024, Hz = 16, HDz = 64;
constexpr int BT = Bz * Tz;       // 4096 rows
constexpr int FF = 4 * Ez;        // 4096

#if __has_builtin(__builtin_amdgcn_global_load_async_to_lds_b128)
#define HAVE_ASYNC_LDS 1
#endif

// ---- WMMA helper -----------------------------------------------------------
__device__ __forceinline__ f32x8 wmma_bf16(bf16x16 a, bf16x16 b, f32x8 c) {
  return __builtin_amdgcn_wmma_f32_16x16x32_bf16(
      /*neg_a=*/false, a, /*neg_b=*/false, b,
      /*c_mod=*/(short)0, c, /*reuse_a=*/false, /*reuse_b=*/false);
}

// A-fragment (16x32, 16-bit): lane holds row = lane%16.
// lanes 0-15: elems 0-7 -> K 0..7,  elems 8-15 -> K 16..23
// lanes16-31: elems 0-7 -> K 8..15, elems 8-15 -> K 24..31
__device__ __forceinline__ bf16x16 load_fragA(const bf16* __restrict__ base,
                                              int stride) {
  int lane = threadIdx.x & 31;
  int row  = lane & 15;
  int koff = (lane & 16) ? 8 : 0;
  const bf16* p = base + (size_t)row * stride + koff;
  bf16x8 lo = *(const bf16x8*)(p);
  bf16x8 hi = *(const bf16x8*)(p + 16);
  bf16x16 r;
#pragma unroll
  for (int i = 0; i < 8; ++i) { r[i] = lo[i]; r[i + 8] = hi[i]; }
  return r;
}

// B-fragment (32x16, 16-bit): lane holds col = lane%16.
// lanes 0-15: elems 0-7 -> K 0..7,  elems 8-15 -> K 16..23
// lanes16-31: elems 0-7 -> K 8..15, elems 8-15 -> K 24..31
// Memory layout Bt[col][k] (K-contiguous per column, `stride` elems per col).
__device__ __forceinline__ bf16x16 load_fragB(const bf16* __restrict__ base,
                                              int stride) {
  int lane = threadIdx.x & 31;
  const bf16* p = base + (size_t)(lane & 15) * stride + ((lane & 16) ? 8 : 0);
  bf16x8 lo = *(const bf16x8*)(p);
  bf16x8 hi = *(const bf16x8*)(p + 16);
  bf16x16 r;
#pragma unroll
  for (int i = 0; i < 8; ++i) { r[i] = lo[i]; r[i + 8] = hi[i]; }
  return r;
}

// ---- async global -> LDS copy of 16 bytes (per lane) -----------------------
__device__ __forceinline__ void async_copy16(const bf16* g, bf16* l) {
#ifdef HAVE_ASYNC_LDS
  __builtin_amdgcn_global_load_async_to_lds_b128((as1_int4p)g, (as3_int4p)l,
                                                 0, 0);
#else
  *(bf16x8*)l = *(const bf16x8*)g;   // synchronous fallback
#endif
}
__device__ __forceinline__ void async_wait_keep2() {
#ifdef HAVE_ASYNC_LDS
  asm volatile("s_wait_asynccnt 0x2" ::: "memory");
#endif
}
__device__ __forceinline__ void async_wait_all() {
#ifdef HAVE_ASYNC_LDS
  asm volatile("s_wait_asynccnt 0x0" ::: "memory");
#endif
}

// ---- Weight transpose + fp32->bf16 cast: Wt[n][k] = bf16(W[k][n]) ----------
__global__ __launch_bounds__(256) void transpose_cast(
    const float* __restrict__ W, bf16* __restrict__ Wt, int K, int N) {
  __shared__ float tile[32][33];
  int bk = blockIdx.y * 32, bn = blockIdx.x * 32;
  int tx = threadIdx.x & 31, ty = threadIdx.x >> 5;  // 32 x 8
#pragma unroll
  for (int i = 0; i < 32; i += 8)
    tile[ty + i][tx] = W[(size_t)(bk + ty + i) * N + (bn + tx)];
  __syncthreads();
#pragma unroll
  for (int i = 0; i < 32; i += 8)
    Wt[(size_t)(bn + ty + i) * K + (bk + tx)] = (bf16)tile[tx][ty + i];
}

// ---- LayerNorm (fp32 in -> bf16 out), one row (E=1024) per 256-thr block ---
__global__ __launch_bounds__(256) void layernorm_bf16(
    const float* __restrict__ X, const float* __restrict__ g,
    const float* __restrict__ be, bf16* __restrict__ Y) {
  __shared__ float red[2][8];
  int row = blockIdx.x;
  int tid = threadIdx.x, w = tid >> 5, lane = tid & 31;
  const float* x = X + (size_t)row * Ez;
  float4 v = ((const float4*)x)[tid];            // 256 * 4 = 1024
  float s  = v.x + v.y + v.z + v.w;
  float ss = v.x * v.x + v.y * v.y + v.z * v.z + v.w * v.w;
#pragma unroll
  for (int off = 1; off < 32; off <<= 1) {
    s  += __shfl_xor(s,  off, 32);
    ss += __shfl_xor(ss, off, 32);
  }
  if (lane == 0) { red[0][w] = s; red[1][w] = ss; }
  __syncthreads();
  float st = 0.f, sst = 0.f;
#pragma unroll
  for (int i = 0; i < 8; ++i) { st += red[0][i]; sst += red[1][i]; }
  float mu  = st * (1.0f / Ez);
  float var = sst * (1.0f / Ez) - mu * mu;
  float inv = rsqrtf(var + 1e-5f);
  float4 gg = ((const float4*)g)[tid];
  float4 bb = ((const float4*)be)[tid];
  bf16x4 o;
  o[0] = (bf16)((v.x - mu) * inv * gg.x + bb.x);
  o[1] = (bf16)((v.y - mu) * inv * gg.y + bb.y);
  o[2] = (bf16)((v.z - mu) * inv * gg.z + bb.z);
  o[3] = (bf16)((v.w - mu) * inv * gg.w + bb.w);
  ((bf16x4*)(Y + (size_t)row * Ez))[tid] = o;
}

// ---- Generic WMMA GEMM: C[M,N] = A[M,K](bf16,row) * Bt[N,K](bf16,K-major) --
// mode 0: scatter bf16 to [B,H,T,HD]        (Q/K; scale folds HD^-0.5 into Q)
// mode 1: scatter bf16 to [B,H,HD,T]        (V transposed)
// mode 2: outF[m,n] = res[m,n] + acc + bias[n]              (fp32)
// mode 3: outB[m,n] = bf16(relu(acc + bias[n]))             (bf16)
__global__ __launch_bounds__(256) void gemm_bf16(
    const bf16* __restrict__ A, const bf16* __restrict__ Bt,
    int M, int N, int K, int mode, float scale,
    const float* __restrict__ bias, const float* __restrict__ res,
    float* __restrict__ outF, bf16* __restrict__ outB) {
  int w = threadIdx.x >> 5, lane = threadIdx.x & 31;
  int wm = w & 3, wn = w >> 2;                     // 4 x 2 waves
  int row0 = blockIdx.x * 128 + wm * 32;
  int col0 = blockIdx.y * 64 + wn * 32;
  f32x8 acc[2][2] = {};
  for (int k = 0; k < K; k += 32) {
    __builtin_prefetch(A + (size_t)row0 * K + k + 128, 0, 3);
    __builtin_prefetch(Bt + (size_t)col0 * K + k + 128, 0, 3);
    bf16x16 a0 = load_fragA(A + (size_t)row0 * K + k, K);
    bf16x16 a1 = load_fragA(A + (size_t)(row0 + 16) * K + k, K);
    bf16x16 b0 = load_fragB(Bt + (size_t)col0 * K + k, K);
    bf16x16 b1 = load_fragB(Bt + (size_t)(col0 + 16) * K + k, K);
    acc[0][0] = wmma_bf16(a0, b0, acc[0][0]);
    acc[0][1] = wmma_bf16(a0, b1, acc[0][1]);
    acc[1][0] = wmma_bf16(a1, b0, acc[1][0]);
    acc[1][1] = wmma_bf16(a1, b1, acc[1][1]);
  }
  int moff = (lane & 16) ? 8 : 0;
  int ncol = lane & 15;
#pragma unroll
  for (int mi = 0; mi < 2; ++mi)
#pragma unroll
    for (int ni = 0; ni < 2; ++ni)
#pragma unroll
      for (int i = 0; i < 8; ++i) {
        int m = row0 + mi * 16 + moff + i;
        int n = col0 + ni * 16 + ncol;
        float v = acc[mi][ni][i] * scale;
        if (mode == 0) {
          int b = m / Tz, t = m % Tz, hh = n / HDz, d = n % HDz;
          outB[(((size_t)(b * Hz + hh)) * Tz + t) * HDz + d] = (bf16)v;
        } else if (mode == 1) {
          int b = m / Tz, t = m % Tz, hh = n / HDz, d = n % HDz;
          outB[(((size_t)(b * Hz + hh)) * HDz + d) * Tz + t] = (bf16)v;
        } else if (mode == 2) {
          outF[(size_t)m * N + n] = res[(size_t)m * N + n] + v + bias[n];
        } else {
          float u = v + bias[n];
          outB[(size_t)m * N + n] = (bf16)(u > 0.f ? u : 0.f);
        }
      }
}

// ---- Flash attention, block-cooperative K/V staging ------------------------
// 8 waves/block = 8 consecutive 16-query tiles of the SAME (b,h).
// Per 32-key block: K tile (32x64 bf16) + Vt tile (64x32 bf16) are async-
// copied to LDS once (1 b128/lane each), double buffered, shared by all waves.
// q,k: [B,H,T,HD] bf16 (q pre-scaled by HD^-0.5), vt: [B,H,HD,T] bf16.
__global__ __launch_bounds__(256) void flash_attn(
    const bf16* __restrict__ q, const bf16* __restrict__ k,
    const bf16* __restrict__ vt, bf16* __restrict__ attn) {
  __shared__ __attribute__((aligned(16))) bf16 pbuf[8][16 * 48];  // P tiles
  __shared__ __attribute__((aligned(16))) bf16 kT[2][32 * 64];    // [key][d]
  __shared__ __attribute__((aligned(16))) bf16 vT[2][64 * 32];    // [d][t]
  int tid = threadIdx.x, w = tid >> 5, lane = tid & 31;
  const int QT = Tz / 16;              // 128 query tiles per (b,h)
  const int BPB = QT / 8;              // 16 blocks per (b,h)
  int bh   = blockIdx.x / BPB;
  int qblk = blockIdx.x % BPB;
  int q0   = (qblk * 8 + w) * 16;      // this wave's query tile
  const bf16* qb = q  + (size_t)bh * Tz * HDz;
  const bf16* kb = k  + (size_t)bh * Tz * HDz;
  const bf16* vb = vt + (size_t)bh * HDz * Tz;
  bf16* pT = pbuf[w];

  bf16x16 qa0 = load_fragA(qb + (size_t)q0 * HDz, HDz);
  bf16x16 qa1 = load_fragA(qb + (size_t)q0 * HDz + 32, HDz);

  f32x8 O[4] = {};
  float mrow[8], lrow[8];
#pragma unroll
  for (int i = 0; i < 8; ++i) { mrow[i] = -1e30f; lrow[i] = 0.f; }
  int moff = (lane & 16) ? 8 : 0;
  int ncol = lane & 15;

  int nkb    = (q0 + 47) / 32;                       // my causal range
  int nkbMax = ((qblk * 8 + 7) * 16 + 47) / 32;      // block's range (wave 7)

  // cooperative tile fetch: K tile is 4KB contiguous; Vt tile is 64 rows x 64B
  int vrow = tid >> 2, vseg = (tid & 3) * 8;
  auto stage = [&](int buf, int t0) {
    async_copy16(kb + (size_t)t0 * HDz + tid * 8, &kT[buf][tid * 8]);
    async_copy16(vb + (size_t)vrow * Tz + t0 + vseg, &vT[buf][vrow * 32 + vseg]);
  };

  stage(0, 0);
  for (int kbI = 0; kbI < nkbMax; ++kbI) {
    int cur = kbI & 1;
    bool more = (kbI + 1 < nkbMax);
    if (more) {
      stage(cur ^ 1, (kbI + 1) * 32);   // prefetch next tiles
      async_wait_keep2();               // current tiles done, next in flight
    } else {
      async_wait_all();
    }
    __syncthreads();                    // all lanes' LDS writes visible

    if (kbI < nkb) {
      int t0 = kbI * 32;
      const bf16* kt  = kT[cur];
      const bf16* vtl = vT[cur];
      f32x8 S0 = {}, S1 = {};
      {  // S = Q * K^T over HD=64 (two K-steps of 32), from LDS
        bf16x16 b00 = load_fragB(kt, 64);
        bf16x16 b10 = load_fragB(kt + 16 * 64, 64);
        S0 = wmma_bf16(qa0, b00, S0);
        S1 = wmma_bf16(qa0, b10, S1);
        bf16x16 b01 = load_fragB(kt + 32, 64);
        bf16x16 b11 = load_fragB(kt + 16 * 64 + 32, 64);
        S0 = wmma_bf16(qa1, b01, S0);
        S1 = wmma_bf16(qa1, b11, S1);
      }
#pragma unroll
      for (int i = 0; i < 8; ++i) {
        int qrow = q0 + moff + i;
        float s0 = (t0 + ncol      <= qrow) ? S0[i] : -1e30f;
        float s1 = (t0 + 16 + ncol <= qrow) ? S1[i] : -1e30f;
        float vmax = fmaxf(s0, s1);
#pragma unroll
        for (int off = 1; off < 16; off <<= 1)
          vmax = fmaxf(vmax, __shfl_xor(vmax, off, 32));
        float mnew  = fmaxf(mrow[i], vmax);
        float alpha = __expf(mrow[i] - mnew);
        float p0 = __expf(s0 - mnew), p1 = __expf(s1 - mnew);
        float rs = p0 + p1;
#pragma unroll
        for (int off = 1; off < 16; off <<= 1) rs += __shfl_xor(rs, off, 32);
        lrow[i] = lrow[i] * alpha + rs;
        mrow[i] = mnew;
#pragma unroll
        for (int c = 0; c < 4; ++c) O[c][i] *= alpha;
        int prow = moff + i;
        pT[prow * 48 + ncol]      = (bf16)p0;
        pT[prow * 48 + 16 + ncol] = (bf16)p1;
      }
      asm volatile("s_wait_dscnt 0" ::: "memory");   // P stores visible (wave)
      bf16x16 pa = load_fragA(pT, 48);               // P as A-fragment
#pragma unroll
      for (int c = 0; c < 4; ++c) {
        bf16x16 bv = load_fragB(vtl + c * 16 * 32, 32);
        O[c] = wmma_bf16(pa, bv, O[c]);
      }
    }
    __syncthreads();                    // done reading cur before reuse
  }

  int hh = bh % Hz, b = bh / Hz;
#pragma unroll
  for (int c = 0; c < 4; ++c)
#pragma unroll
    for (int i = 0; i < 8; ++i) {
      int t = q0 + moff + i;
      int col = hh * HDz + c * 16 + ncol;
      attn[((size_t)(b * Tz + t)) * Ez + col] = (bf16)(O[c][i] / lrow[i]);
    }
}

// ---------------------------------------------------------------------------
extern "C" void kernel_launch(void* const* d_in, const int* in_sizes, int n_in,
                              void* d_out, int out_size, void* d_ws,
                              size_t ws_size, hipStream_t stream) {
  const float* x   = (const float*)d_in[0];
  const float* Wk  = (const float*)d_in[1];
  const float* Wq  = (const float*)d_in[2];
  const float* Wv  = (const float*)d_in[3];
  const float* Wp  = (const float*)d_in[4];
  const float* bp  = (const float*)d_in[5];
  const float* W1  = (const float*)d_in[6];
  const float* b1  = (const float*)d_in[7];
  const float* W2  = (const float*)d_in[8];
  const float* b2  = (const float*)d_in[9];
  const float* g1  = (const float*)d_in[10];
  const float* be1 = (const float*)d_in[11];
  const float* g2  = (const float*)d_in[12];
  const float* be2 = (const float*)d_in[13];
  float* out = (float*)d_out;

  char* ws = (char*)d_ws;
  const size_t MB = 1024 * 1024;
  bf16* wqT = (bf16*)(ws + 0 * MB);    // [E,E]  K-major, 2 MB
  bf16* wkT = (bf16*)(ws + 2 * MB);
  bf16* wvT = (bf16*)(ws + 4 * MB);
  bf16* wpT = (bf16*)(ws + 6 * MB);
  bf16* w1T = (bf16*)(ws + 8 * MB);    // [4E,E] 8 MB
  bf16* w2T = (bf16*)(ws + 16 * MB);   // [E,4E] 8 MB
  bf16* hB  = (bf16*)(ws + 24 * MB);   // LN out [B,T,E] bf16 (reused for LN2)
  bf16* qB  = (bf16*)(ws + 32 * MB);   // [B,H,T,HD]
  bf16* kB  = (bf16*)(ws + 40 * MB);   // [B,H,T,HD]
  bf16* vtB = (bf16*)(ws + 48 * MB);   // [B,H,HD,T]
  bf16* atB = (bf16*)(ws + 56 * MB);   // attn out [B,T,E] bf16
  float* x1 = (float*)(ws + 64 * MB);  // residual-1 fp32, 16 MB
  bf16* ff1 = (bf16*)(ws + 32 * MB);   // [B,T,4E] bf16, reuses dead q/k/v/attn

  dim3 blk(256);

  // Weights -> bf16, K-major
  transpose_cast<<<dim3(Ez / 32, Ez / 32), blk, 0, stream>>>(Wq, wqT, Ez, Ez);
  transpose_cast<<<dim3(Ez / 32, Ez / 32), blk, 0, stream>>>(Wk, wkT, Ez, Ez);
  transpose_cast<<<dim3(Ez / 32, Ez / 32), blk, 0, stream>>>(Wv, wvT, Ez, Ez);
  transpose_cast<<<dim3(Ez / 32, Ez / 32), blk, 0, stream>>>(Wp, wpT, Ez, Ez);
  transpose_cast<<<dim3(FF / 32, Ez / 32), blk, 0, stream>>>(W1, w1T, Ez, FF);
  transpose_cast<<<dim3(Ez / 32, FF / 32), blk, 0, stream>>>(W2, w2T, FF, Ez);

  // LN1
  layernorm_bf16<<<dim3(BT), blk, 0, stream>>>(x, g1, be1, hB);

  // QKV projections (Q pre-scaled by HD^-0.5 = 0.125)
  gemm_bf16<<<dim3(BT / 128, Ez / 64), blk, 0, stream>>>(
      hB, wqT, BT, Ez, Ez, 0, 0.125f, nullptr, nullptr, nullptr, qB);
  gemm_bf16<<<dim3(BT / 128, Ez / 64), blk, 0, stream>>>(
      hB, wkT, BT, Ez, Ez, 0, 1.0f, nullptr, nullptr, nullptr, kB);
  gemm_bf16<<<dim3(BT / 128, Ez / 64), blk, 0, stream>>>(
      hB, wvT, BT, Ez, Ez, 1, 1.0f, nullptr, nullptr, nullptr, vtB);

  // Causal flash attention (block-cooperative K/V staging)
  flash_attn<<<dim3(Bz * Hz * (Tz / 16) / 8), blk, 0, stream>>>(qB, kB, vtB,
                                                                atB);

  // Output projection + residual -> x1 (fp32)
  gemm_bf16<<<dim3(BT / 128, Ez / 64), blk, 0, stream>>>(
      atB, wpT, BT, Ez, Ez, 2, 1.0f, bp, x, x1, nullptr);

  // LN2
  layernorm_bf16<<<dim3(BT), blk, 0, stream>>>(x1, g2, be2, hB);

  // FFN
  gemm_bf16<<<dim3(BT / 128, FF / 64), blk, 0, stream>>>(
      hB, w1T, BT, FF, Ez, 3, 1.0f, b1, nullptr, nullptr, ff1);
  gemm_bf16<<<dim3(BT / 128, Ez / 64), blk, 0, stream>>>(
      ff1, w2T, BT, Ez, FF, 2, 1.0f, b2, x1, out, nullptr);
}
